// GotenNetCore_9182640078980
// MI455X (gfx1250) — matrix operand order repaired and checked by reading, not compiled, not run on hardware
//
#include <hip/hip_runtime.h>
#include <math.h>

#define NN 20000
#define EE 320000
#define HH 128
#define LL 6
#define RBFN 32
#define CUT 5.0f

typedef __attribute__((ext_vector_type(16))) __bf16 v16bf;
typedef __attribute__((ext_vector_type(8)))  float  v8f;

// ---------- helpers ----------
static __device__ __forceinline__ unsigned short f2bf(float x) {
  union { __bf16 b; unsigned short u; } v;
  v.b = (__bf16)x;                      // native v_cvt (RNE)
  return v.u;
}
static __device__ __forceinline__ float bf2f(unsigned short h) {
  union { unsigned short u; __bf16 b; } v;
  v.u = h;
  return (float)v.b;
}
static __device__ __forceinline__ float silu_f(float x) {
  return x / (1.0f + __expf(-x));
}
static __device__ __forceinline__ v8f wmma_bf16(v16bf a, v16bf b, v8f c) {
  // D = A(16x32 bf16) * B(32x16 bf16) + C(16x16 f32)
  return __builtin_amdgcn_wmma_f32_16x16x32_bf16(false, a, false, b, (short)0, c, false, false);
}
// A fragment (16x32) from a row-major bf16 tile (LDS). Layout:
// lanes 0-15 row r: VGPR j -> K=16*hi+2j (contiguous, matches load_B_tr)
static __device__ __forceinline__ v16bf load_A_rm(const unsigned short* base, int stride,
                                                  int kkOff, int lane) {
  int r = lane & 15, hi = (lane >> 4) & 1;
  const unsigned short* row = base + r * stride + kkOff;
  union { v16bf v; unsigned int u[8]; } f;
#pragma unroll
  for (int j = 0; j < 8; ++j) {
    int k0 = 16 * hi + 2 * j;
    f.u[j] = *(const unsigned int*)(row + k0);
  }
  return f.v;
}
// B fragment (32x16) from transposed weight Wt[col][k] (bf16, row length Klen):
// lane col = lane&15, K half = lane>>4 -> 16 contiguous bf16 = 2x b128 loads.
static __device__ __forceinline__ v16bf load_B_tr(const unsigned short* Wt, int colBase,
                                                  int Klen, int kkOff, int lane) {
  int col = lane & 15, khalf = (lane >> 4) & 1;
  const unsigned short* p = Wt + (size_t)(colBase + col) * Klen + kkOff + khalf * 16;
  union { v16bf v; uint4 q[2]; } f;
  f.q[0] = *(const uint4*)(p);
  f.q[1] = *(const uint4*)(p + 8);
  return f.v;
}
struct F3 { float x, y, z; };   // 12B vector gather -> global_load_b96

// ---------- prep kernels ----------
__global__ void k_embed(const int* __restrict__ z, const float* __restrict__ emb,
                        float* __restrict__ buf) {
  int i = blockIdx.x * 256 + threadIdx.x;          // i < N*512
  int n = i >> 9, r = i & 511;
  if (r < HH) buf[(size_t)n * HH + r] = emb[(size_t)z[n] * HH + r];
  else        buf[(size_t)NN * HH + (size_t)n * 384 + (r - HH)] = 0.0f;
}

__global__ void k_edgeprep(const float* __restrict__ ediff, const float* __restrict__ evec,
                           unsigned short* __restrict__ ea, float* __restrict__ unit) {
  int e = blockIdx.x * 256 + threadIdx.x;
  if (e >= EE) return;
  float d = ediff[e];
  const float step  = CUT / (RBFN - 1);
  const float coeff = -0.5f / (step * step);
  float cut = 0.5f * (cosf(d * (3.14159265358979f / CUT)) + 1.0f) * (d < CUT ? 1.0f : 0.0f);
#pragma unroll
  for (int k = 0; k < RBFN; ++k) {
    float t = d - k * step;
    ea[(size_t)e * RBFN + k] = f2bf(__expf(coeff * t * t) * cut);
  }
  float inv = 1.0f / (d + 1e-6f);
  unit[e * 3 + 0] = evec[e * 3 + 0] * inv;
  unit[e * 3 + 1] = evec[e * 3 + 1] * inv;
  unit[e * 3 + 2] = evec[e * 3 + 2] * inv;
}

__global__ void k_transpose(const float* __restrict__ in, unsigned short* __restrict__ out,
                            int R, int C) {
  int i = blockIdx.x * 256 + threadIdx.x;
  if (i >= R * C) return;
  int r = i / C, c = i % C;
  out[(size_t)c * R + r] = f2bf(in[i]);
}

// ---------- fused edge interaction: edge-MLP (WMMA) + message + scatter ----------
__global__ __launch_bounds__(256) void k_interact(
    const float* __restrict__ s_cur, const float* __restrict__ v_cur,
    float* __restrict__ s_next, float* __restrict__ v_next,
    const int* __restrict__ src, const int* __restrict__ dst,
    const float* __restrict__ unit, const unsigned short* __restrict__ ea,
    const unsigned short* __restrict__ W1t, const float* __restrict__ b1,
    const unsigned short* __restrict__ W2t, const float* __restrict__ b2) {
  __shared__ __align__(16) unsigned short sh_ea[128 * 32];   // 8 KB
  __shared__ __align__(16) unsigned short sh_h[128 * 128];   // 32 KB
  __shared__ int   sh_src[128], sh_dst[128];
  __shared__ float sh_unit[128 * 3];

  int tid = threadIdx.x;
  int eb  = blockIdx.x * 128;

  // Async copy of the edge-feature tile: global -> LDS (ASYNCcnt path)
  {
    unsigned lBase = (unsigned)(size_t)(&sh_ea[0]);           // wave-relative LDS offset
    const char* gBase = (const char*)(ea + (size_t)eb * 32);  // 8 KB tile, 16B aligned
#pragma unroll
    for (int it = 0; it < 2; ++it) {
      int chunk = tid + it * 256;                             // 512 x 16B chunks
      unsigned loff = lBase + chunk * 16;
      unsigned long long ga = (unsigned long long)(size_t)(gBase + chunk * 16);
      asm volatile("global_load_async_to_lds_b128 %0, %1, off"
                   :: "v"(loff), "v"(ga) : "memory");
    }
  }
  for (int i = tid; i < 128; i += 256) {
    sh_src[i] = src[eb + i];
    sh_dst[i] = dst[eb + i];
    sh_unit[i * 3 + 0] = unit[(size_t)(eb + i) * 3 + 0];
    sh_unit[i * 3 + 1] = unit[(size_t)(eb + i) * 3 + 1];
    sh_unit[i * 3 + 2] = unit[(size_t)(eb + i) * 3 + 2];
  }
  asm volatile("s_wait_asynccnt 0" ::: "memory");
  __syncthreads();

  int w = tid >> 5, lane = tid & 31;
  int col = lane & 15, hi = (lane >> 4) & 1;

  // Stage 1: h = silu(ea @ W1 + b1), one 16x16 tile per (wave, n)
  {
    v16bf a = load_A_rm(sh_ea + w * 16 * 32, 32, 0, lane);
#pragma unroll
    for (int n = 0; n < 8; ++n) {
      v16bf b = load_B_tr(W1t, n * 16, 32, 0, lane);
      v8f acc = {};
      acc = wmma_bf16(a, b, acc);
      float bias = b1[n * 16 + col];
#pragma unroll
      for (int j = 0; j < 8; ++j) {
        float x = acc[j] + bias;
        sh_h[(w * 16 + j + 8 * hi) * 128 + n * 16 + col] = f2bf(silu_f(x));
      }
    }
  }
  __syncthreads();

  // Stage 2: W = h @ W2 + b2 (W_s | W_v1 | W_v2), fused message + scatter-add.
  // Hoist the 4 K-chunk A fragments once; reuse across all 8 channel tiles.
  v16bf aF[4];
#pragma unroll
  for (int kk = 0; kk < 4; ++kk)
    aF[kk] = load_A_rm(sh_h + w * 16 * 128, 128, kk * 32, lane);

  for (int c = 0; c < 8; ++c) {
    v8f as = {}, a1 = {}, a2 = {};
#pragma unroll
    for (int kk = 0; kk < 4; ++kk) {
      v16bf bs  = load_B_tr(W2t,        c * 16, 128, kk * 32, lane);
      v16bf bv1 = load_B_tr(W2t, 128 +  c * 16, 128, kk * 32, lane);
      v16bf bv2 = load_B_tr(W2t, 256 +  c * 16, 128, kk * 32, lane);
      as = wmma_bf16(aF[kk], bs,  as);
      a1 = wmma_bf16(aF[kk], bv1, a1);
      a2 = wmma_bf16(aF[kk], bv2, a2);
    }
    int ch = c * 16 + col;
    float bS = b2[ch], bV1 = b2[128 + ch], bV2 = b2[256 + ch];
#pragma unroll
    for (int j = 0; j < 8; ++j) {
      int el = w * 16 + j + 8 * hi;
      int sn = sh_src[el], dn = sh_dst[el];
      float sj  = s_cur[(size_t)sn * 128 + ch];
      F3 vj = *(const F3*)(v_cur + (size_t)sn * 384 + ch * 3);  // b96 gather
      float Ws  = as[j] + bS, Wv1 = a1[j] + bV1, Wv2 = a2[j] + bV2;
      unsafeAtomicAdd(&s_next[(size_t)dn * 128 + ch], sj * Ws);
      float svw = sj * Wv2;
      float* vd = v_next + (size_t)dn * 384 + ch * 3;
      unsafeAtomicAdd(vd + 0, vj.x * Wv1 + svw * sh_unit[el * 3 + 0]);
      unsafeAtomicAdd(vd + 1, vj.y * Wv1 + svw * sh_unit[el * 3 + 1]);
      unsafeAtomicAdd(vd + 2, vj.z * Wv1 + svw * sh_unit[el * 3 + 2]);
    }
  }
}

// ---------- fused node update (all 5 GEMMs in WMMA) ----------
__global__ __launch_bounds__(256) void k_update(
    float* __restrict__ sbuf,                       // s at sbuf, v at sbuf + N*128
    const unsigned short* __restrict__ Wvpt,
    const unsigned short* __restrict__ U1t, const float* __restrict__ ub1,
    const unsigned short* __restrict__ U2t, const float* __restrict__ ub2,
    const unsigned short* __restrict__ Wst, const float* __restrict__ ubs) {
  __shared__ __align__(16) unsigned short sh_vbf[3 * 16 * 128];  // 12 KB [d][n][c]
  __shared__ __align__(16) unsigned short sh_vu [3 * 16 * 128];  // 12 KB
  __shared__ __align__(16) unsigned short sh_vv [3 * 16 * 128];  // 12 KB
  __shared__ __align__(16) unsigned short sh_feat[16 * 256];     // 8 KB
  __shared__ __align__(16) unsigned short sh_x1 [16 * 128];      // 4 KB
  __shared__ float          sh_avv[16 * 128];                    // 8 KB
  __shared__ __align__(16) unsigned short sh_t2 [16 * 128];      // 4 KB

  int tid = threadIdx.x;
  int nb  = blockIdx.x * 16;
  int w = tid >> 5, lane = tid & 31;
  int col = lane & 15, hi = (lane >> 4) & 1;
  float* vbuf = sbuf + (size_t)NN * 128;

  // stage v into bf16 LDS, per-direction row-major slices
  for (int i = tid; i < 6144; i += 256) {
    int n = i / 384, rem = i % 384, c = rem / 3, d = rem % 3;
    sh_vbf[d * 2048 + n * 128 + c] = f2bf(vbuf[(size_t)nb * 384 + i]);
  }
  __syncthreads();

  // v_proj = v @ Wvp  -> v_u (k<128) / v_v (k>=128), 48 tiles over 8 waves
  for (int t = w; t < 48; t += 8) {
    int d = t >> 4, kt = t & 15;
    v8f acc = {};
#pragma unroll
    for (int kk = 0; kk < 4; ++kk) {
      v16bf a = load_A_rm(sh_vbf + d * 2048, 128, kk * 32, lane);
      v16bf b = load_B_tr(Wvpt, kt * 16, 128, kk * 32, lane);
      acc = wmma_bf16(a, b, acc);
    }
#pragma unroll
    for (int j = 0; j < 8; ++j) {
      int m = j + 8 * hi, kg = kt * 16 + col;
      unsigned short bv = f2bf(acc[j]);
      if (kt < 8) sh_vu[d * 2048 + m * 128 + kg]        = bv;
      else        sh_vv[d * 2048 + m * 128 + (kg - 128)] = bv;
    }
  }
  __syncthreads();

  // feat = [s, ||v_v||]
  for (int i = tid; i < 4096; i += 256) {
    int n = i >> 8, c = i & 255;
    float val;
    if (c < 128) val = sbuf[(size_t)(nb + n) * 128 + c];
    else {
      int cc = c - 128;
      float a2 = 0.0f;
#pragma unroll
      for (int d = 0; d < 3; ++d) {
        float x = bf2f(sh_vv[d * 2048 + n * 128 + cc]);
        a2 += x * x;
      }
      val = sqrtf(a2);
    }
    sh_feat[n * 256 + c] = f2bf(val);
  }
  __syncthreads();

  // x1 = silu(feat @ U1 + b1), wave w owns column tile w (K=256)
  {
    v8f acc = {};
#pragma unroll
    for (int kk = 0; kk < 8; ++kk) {
      v16bf a = load_A_rm(sh_feat, 256, kk * 32, lane);
      v16bf b = load_B_tr(U1t, w * 16, 256, kk * 32, lane);
      acc = wmma_bf16(a, b, acc);
    }
    float bias = ub1[w * 16 + col];
#pragma unroll
    for (int j = 0; j < 8; ++j)
      sh_x1[(j + 8 * hi) * 128 + w * 16 + col] = f2bf(silu_f(acc[j] + bias));
  }
  __syncthreads();

  // s_out = x1 @ U2 + b2 -> a_vv|a_sv|a_ss, fuse inner-product gate -> t2
  {
    v8f av = {}, asv = {}, ass = {};
#pragma unroll
    for (int kk = 0; kk < 4; ++kk) {
      v16bf a   = load_A_rm(sh_x1, 128, kk * 32, lane);
      v16bf b0  = load_B_tr(U2t,        w * 16, 128, kk * 32, lane);
      v16bf b1v = load_B_tr(U2t, 128 + w * 16, 128, kk * 32, lane);
      v16bf b2v = load_B_tr(U2t, 256 + w * 16, 128, kk * 32, lane);
      av  = wmma_bf16(a, b0,  av);
      asv = wmma_bf16(a, b1v, asv);
      ass = wmma_bf16(a, b2v, ass);
    }
    int cg = w * 16 + col;
    float bbv = ub2[cg], bbs = ub2[128 + cg], bbS = ub2[256 + cg];
#pragma unroll
    for (int j = 0; j < 8; ++j) {
      int m = j + 8 * hi;
      float a_vv = av[j] + bbv, a_sv = asv[j] + bbs, a_ss = ass[j] + bbS;
      float inner = 0.0f;
#pragma unroll
      for (int d = 0; d < 3; ++d)
        inner += bf2f(sh_vu[d * 2048 + m * 128 + cg]) * bf2f(sh_vv[d * 2048 + m * 128 + cg]);
      sh_avv[m * 128 + cg] = a_vv;
      sh_t2[m * 128 + cg]  = f2bf(a_ss + inner * a_sv);
    }
  }
  __syncthreads();

  // s += t2 @ Ws + bs
  {
    v8f acc = {};
#pragma unroll
    for (int kk = 0; kk < 4; ++kk) {
      v16bf a = load_A_rm(sh_t2, 128, kk * 32, lane);
      v16bf b = load_B_tr(Wst, w * 16, 128, kk * 32, lane);
      acc = wmma_bf16(a, b, acc);
    }
    float bias = ubs[w * 16 + col];
#pragma unroll
    for (int j = 0; j < 8; ++j) {
      int m = j + 8 * hi;
      size_t idx = (size_t)(nb + m) * 128 + w * 16 + col;
      sbuf[idx] = sbuf[idx] + acc[j] + bias;
    }
  }
  // v += v_u * a_vv
  for (int i = tid; i < 6144; i += 256) {
    int n = i / 384, rem = i % 384, c = rem / 3, d = rem % 3;
    size_t gi = (size_t)nb * 384 + i;
    vbuf[gi] = vbuf[gi] + bf2f(sh_vu[d * 2048 + n * 128 + c]) * sh_avv[n * 128 + c];
  }
}

// ---------- host ----------
extern "C" void kernel_launch(void* const* d_in, const int* in_sizes, int n_in,
                              void* d_out, int out_size, void* d_ws, size_t ws_size,
                              hipStream_t stream) {
  (void)in_sizes; (void)n_in; (void)out_size; (void)ws_size;
  const int*   z     = (const int*)d_in[0];
  const int*   eidx  = (const int*)d_in[2];
  const float* evec  = (const float*)d_in[3];
  const float* ediff = (const float*)d_in[4];
  const float* emb   = (const float*)d_in[5];
  const float* iW1   = (const float*)d_in[6];
  const float* ib1   = (const float*)d_in[7];
  const float* iW2   = (const float*)d_in[8];
  const float* ib2   = (const float*)d_in[9];
  const float* uWvp  = (const float*)d_in[10];
  const float* uW1   = (const float*)d_in[11];
  const float* ub1   = (const float*)d_in[12];
  const float* uW2   = (const float*)d_in[13];
  const float* ub2   = (const float*)d_in[14];
  const float* uWs   = (const float*)d_in[15];
  const float* ubs   = (const float*)d_in[16];

  char* p = (char*)d_ws;
  float* stateB = (float*)p;            p += (size_t)NN * 512 * 4;   // alt state buffer
  float* unit   = (float*)p;            p += (size_t)EE * 3 * 4;
  unsigned short* ea   = (unsigned short*)p; p += (size_t)EE * 32 * 2;
  unsigned short* w1t  = (unsigned short*)p; p += (size_t)LL * 32 * 128 * 2;
  unsigned short* w2t  = (unsigned short*)p; p += (size_t)LL * 128 * 384 * 2;
  unsigned short* wvpt = (unsigned short*)p; p += (size_t)LL * 128 * 256 * 2;
  unsigned short* u1t  = (unsigned short*)p; p += (size_t)LL * 256 * 128 * 2;
  unsigned short* u2t  = (unsigned short*)p; p += (size_t)LL * 128 * 384 * 2;
  unsigned short* wst  = (unsigned short*)p; p += (size_t)LL * 128 * 128 * 2;

  float* bufA = (float*)d_out;   // initial state lives in d_out; 6 swaps -> ends in d_out
  float* bufB = stateB;

  k_embed<<<(NN * 512) / 256, 256, 0, stream>>>(z, emb, bufA);
  k_edgeprep<<<(EE + 255) / 256, 256, 0, stream>>>(ediff, evec, ea, unit);
  for (int l = 0; l < LL; ++l) {
    k_transpose<<<(32 * 128 + 255) / 256, 256, 0, stream>>>(iW1 + l * 4096,  w1t  + l * 4096,  32, 128);
    k_transpose<<<(128 * 384 + 255) / 256, 256, 0, stream>>>(iW2 + l * 49152, w2t  + l * 49152, 128, 384);
    k_transpose<<<(128 * 256 + 255) / 256, 256, 0, stream>>>(uWvp + l * 32768, wvpt + l * 32768, 128, 256);
    k_transpose<<<(256 * 128 + 255) / 256, 256, 0, stream>>>(uW1 + l * 32768, u1t  + l * 32768, 256, 128);
    k_transpose<<<(128 * 384 + 255) / 256, 256, 0, stream>>>(uW2 + l * 49152, u2t  + l * 49152, 128, 384);
    k_transpose<<<(128 * 128 + 255) / 256, 256, 0, stream>>>(uWs + l * 16384, wst  + l * 16384, 128, 128);
  }

  float* cur = bufA;
  float* nxt = bufB;
  for (int l = 0; l < LL; ++l) {
    hipMemcpyAsync(nxt, cur, (size_t)NN * 512 * 4, hipMemcpyDeviceToDevice, stream);
    k_interact<<<EE / 128, 256, 0, stream>>>(
        cur, cur + (size_t)NN * 128, nxt, nxt + (size_t)NN * 128,
        eidx, eidx + EE, unit, ea,
        w1t + l * 4096, ib1 + l * 128, w2t + l * 49152, ib2 + l * 384);
    k_update<<<NN / 16, 256, 0, stream>>>(
        nxt, wvpt + l * 32768, u1t + l * 32768, ub1 + l * 128,
        u2t + l * 49152, ub2 + l * 384, wst + l * 16384, ubs + l * 128);
    float* t = cur; cur = nxt; nxt = t;
  }
  // cur == d_out here (even number of swaps)
}
